// GCN_69801808495208
// MI455X (gfx1250) — matrix operand order repaired and checked by reading, not compiled, not run on hardware
//
#include <hip/hip_runtime.h>

#define SEQ 128
#define NCH 256
#define NSTEP 3
#define LDS_STRIDE 264   // 264 ushort = 528B rows -> 4-word bank skew, conflict-free tile loads

typedef __attribute__((ext_vector_type(16))) __bf16 v16bf;
typedef __attribute__((ext_vector_type(8)))  float  v8f;

union FragBF {
    v16bf v;
    unsigned short u[16];
};

__device__ __forceinline__ unsigned short f2bf(float f) {
    unsigned int u = __float_as_uint(f);
    u += 0x7FFFu + ((u >> 16) & 1u);   // round-to-nearest-even
    return (unsigned short)(u >> 16);
}
__device__ __forceinline__ float bf2f(unsigned short h) {
    return __uint_as_float(((unsigned int)h) << 16);
}

// One wave computes a 16-row block of  D = A(LDS bf16) @ W.T(global bf16) with
// per-column f32 init; K = N = 256, tiled as 16 column tiles x 8 k-steps of 32.
template <class CInit, class COut>
__device__ __forceinline__ void wave_gemm(const unsigned short* __restrict__ Asrc,
                                          const unsigned short* __restrict__ W,
                                          int m0, int lane, CInit cinit, COut cout) {
    const int msub = lane & 15;
    const int g    = lane >> 4;
    const unsigned short* arow = Asrc + (m0 + msub) * LDS_STRIDE;

    // Hoist all A fragments for this row block (8 k-steps x 8 VGPRs).
    FragBF afr[8];
#pragma unroll
    for (int ks = 0; ks < 8; ++ks) {
        const int k0 = ks * 32;
#pragma unroll
        for (int j = 0; j < 8; ++j) {
            afr[ks].u[j]     = arow[k0 + 8 * g + j];        // VGPR0-3: K = 8g..8g+7
            afr[ks].u[8 + j] = arow[k0 + 16 + 8 * g + j];   // VGPR4-7: K = 16+8g..23+8g
        }
    }

    for (int nt = 0; nt < 16; ++nt) {
        const int n = nt * 16 + msub;
        const float ci = cinit(n);
        v8f acc;
#pragma unroll
        for (int r = 0; r < 8; ++r) acc[r] = ci;

        const unsigned short* wrow = W + n * NCH;   // B[k,n] = W[n,k]: row-contiguous
#pragma unroll
        for (int ks = 0; ks < 8; ++ks) {
            FragBF bfr;
            const int kb = ks * 32 + 8 * g;         // v0-7: K=8g..8g+7; v8-15: K=16+8g..23+8g
#pragma unroll
            for (int v = 0; v < 16; ++v) bfr.u[v] = wrow[kb + (v < 8 ? v : v + 8)];
            acc = __builtin_amdgcn_wmma_f32_16x16x32_bf16(
                false, afr[ks].v, false, bfr.v, (short)0, acc, false, false);
        }
#pragma unroll
        for (int r = 0; r < 8; ++r) cout(m0 + 8 * g + r, n, acc[r]);
    }
}

// Pre-convert all needed weight slices to bf16 in d_ws.
// Layout: [step][mat][256][256], mat: 0=aggW, 1=attnW[:, :256] (W1), 2=updW[:, :256] (U1), 3=updW[:, 256:] (U2)
__global__ void conv_weights(const float* __restrict__ aggW,
                             const float* __restrict__ attnW,
                             const float* __restrict__ updW,
                             unsigned short* __restrict__ wbf) {
    int idx = blockIdx.x * blockDim.x + threadIdx.x;
    if (idx >= NSTEP * 4 * 65536) return;
    int i = idx >> 18;
    int m = (idx >> 16) & 3;
    int r = (idx >> 8) & 255;
    int k = idx & 255;
    float v;
    if (m == 0)      v = aggW[i * 65536 + r * 256 + k];
    else if (m == 1) v = attnW[i * 131072 + r * 512 + k];
    else if (m == 2) v = updW[i * 131072 + r * 512 + k];
    else             v = updW[i * 131072 + r * 512 + 256 + k];
    wbf[idx] = f2bf(v);
}

// One workgroup per batch element; all 3 steps fused. 256 threads = 8 waves.
// Hs holds hid (bf16, pre-masked) and is reused as the p-buffer (hid is dead after phase A).
__global__ __launch_bounds__(256) void gcn_fused(const float* __restrict__ feat,
                                                 const float* __restrict__ mask,
                                                 const float* __restrict__ aggB,
                                                 const float* __restrict__ updB,
                                                 const unsigned short* __restrict__ wbf,
                                                 float* __restrict__ out) {
    __shared__ unsigned short Hs[SEQ * LDS_STRIDE];  // hid / p  (bf16)
    __shared__ unsigned short Xs[SEQ * LDS_STRIDE];  // x        (bf16)
    __shared__ float aggv[NCH];
    __shared__ float vvec[NCH];

    const int b    = blockIdx.x;
    const int tid  = threadIdx.x;
    const int lane = tid & 31;
    const int wv   = tid >> 5;
    const int m0   = wv * 16;

    const float* fb = feat + (size_t)b * SEQ * NCH;
    const float* mb = mask + (size_t)b * SEQ;

    // hid = feat * mask  (mask for step 0 applied at load)
    for (int idx = tid; idx < SEQ * NCH; idx += 256) {
        int s = idx >> 8;
        int c = idx & 255;
        Hs[s * LDS_STRIDE + c] = f2bf(fb[idx] * mb[s]);
    }
    __syncthreads();

    for (int step = 0; step < NSTEP; ++step) {
        const unsigned short* Wa = wbf + (size_t)(step * 4 + 0) * 65536;
        const unsigned short* W1 = wbf + (size_t)(step * 4 + 1) * 65536;
        const unsigned short* U1 = wbf + (size_t)(step * 4 + 2) * 65536;
        const unsigned short* U2 = wbf + (size_t)(step * 4 + 3) * 65536;
        const float* ab = aggB + step * NCH;
        const float* ub = updB + step * NCH;

        // Phase A: X = H @ aggW.T + aggB
        wave_gemm(Hs, Wa, m0, lane,
                  [&](int n) { return ab[n]; },
                  [&](int r, int n, float val) { Xs[r * LDS_STRIDE + n] = f2bf(val); });
        __syncthreads();

        // Phase B: P(=Hs) = X @ W1.T   (q and attnB cancel inside softmax over s)
        wave_gemm(Xs, W1, m0, lane,
                  [&](int) { return 0.0f; },
                  [&](int r, int n, float val) { Hs[r * LDS_STRIDE + n] = f2bf(val); });
        __syncthreads();

        // Phase C: per-channel softmax over s, weighted sum with x, sigmoid.
        {
            const int c = tid;
            float mx = -1e30f;
            for (int s = 0; s < SEQ; ++s)
                mx = fmaxf(mx, bf2f(Hs[s * LDS_STRIDE + c]));
            float den = 0.f, num = 0.f;
            for (int s = 0; s < SEQ; ++s) {
                float e = __expf(bf2f(Hs[s * LDS_STRIDE + c]) - mx);
                den += e;
                num += e * bf2f(Xs[s * LDS_STRIDE + c]);
            }
            float z = num / den;
            aggv[c] = 1.0f / (1.0f + __expf(-z));
        }
        __syncthreads();

        // Phase D1: vvec[n] = agg . U2[n,:] + updB[n]   (tiny vector-matrix product)
        {
            const int n = tid;
            const unsigned short* u2r = U2 + n * NCH;
            float acc = ub[n];
            for (int k = 0; k < NCH; ++k) acc += aggv[k] * bf2f(u2r[k]);
            vvec[n] = acc;
        }
        __syncthreads();

        // Phase D2: hid' = X @ U1.T + vvec (broadcast over rows)
        if (step < NSTEP - 1) {
            wave_gemm(Xs, U1, m0, lane,
                      [&](int n) { return vvec[n]; },
                      [&](int r, int n, float val) {
                          Hs[r * LDS_STRIDE + n] = f2bf(val * mb[r]);  // pre-mask for next step
                      });
        } else {
            float* ob = out + (size_t)b * SEQ * NCH;
            wave_gemm(Xs, U1, m0, lane,
                      [&](int n) { return vvec[n]; },
                      [&](int r, int n, float val) { ob[r * NCH + n] = val; });
        }
        __syncthreads();
    }
}

extern "C" void kernel_launch(void* const* d_in, const int* in_sizes, int n_in,
                              void* d_out, int out_size, void* d_ws, size_t ws_size,
                              hipStream_t stream) {
    const float* feat  = (const float*)d_in[0];
    const float* mask  = (const float*)d_in[1];
    const float* aggW  = (const float*)d_in[2];
    const float* aggB  = (const float*)d_in[3];
    const float* attnW = (const float*)d_in[4];
    // d_in[5] = attnB: cancels inside the softmax, unused
    const float* updW  = (const float*)d_in[6];
    const float* updB  = (const float*)d_in[7];
    float* out = (float*)d_out;
    unsigned short* wbf = (unsigned short*)d_ws;  // 3*4*256*256 bf16 = 1.5 MB

    conv_weights<<<(NSTEP * 4 * 65536 + 255) / 256, 256, 0, stream>>>(aggW, attnW, updW, wbf);
    gcn_fused<<<16, 256, 0, stream>>>(feat, mask, aggB, updB, wbf, out);
}
